// TankBindTriangleSelfAttention_90142773608720
// MI455X (gfx1250) — matrix operand order repaired and checked by reading, not compiled, not run on hardware
//
#include <hip/hip_runtime.h>

// ---------------------------------------------------------------------------
// Types
// ---------------------------------------------------------------------------
typedef __attribute__((ext_vector_type(16))) __bf16 v16bf;
typedef __attribute__((ext_vector_type(8)))  __bf16 v8bf;
typedef __attribute__((ext_vector_type(8)))  float  v8f;
typedef __attribute__((ext_vector_type(4)))  float  v4f;
typedef __attribute__((ext_vector_type(4)))  unsigned int u32x4;
typedef __attribute__((ext_vector_type(8)))  int  i32x8;
typedef __attribute__((ext_vector_type(4)))  int  i32x4;

#define HD __device__ __forceinline__

HD __bf16 f2bf(float f) {
    union { float f; unsigned u; } cv; cv.f = f;
    unsigned u = cv.u;
    u += 0x7fffu + ((u >> 16) & 1u);          // round-to-nearest-even
    union { unsigned short s; __bf16 b; } cb; cb.s = (unsigned short)(u >> 16);
    return cb.b;
}
HD float bf2f(__bf16 b) {
    union { __bf16 b; unsigned short s; } cb; cb.b = b;
    union { unsigned u; float f; } cv; cv.u = ((unsigned)cb.s) << 16;
    return cv.f;
}
HD v8f wmma_bf16(v16bf a, v16bf b, v8f c) {
    return __builtin_amdgcn_wmma_f32_16x16x32_bf16(
        /*neg_a=*/false, a, /*neg_b=*/false, b,
        /*c_mod=*/(short)0, c, /*reuse_a=*/false, /*reuse_b=*/false);
}

union AFrag { v16bf v; v8bf h[2]; };

// Problem constants
#define CC   256
#define HH   8
#define DD   32
#define NN   256

// ---------------------------------------------------------------------------
// Kernel P: pack weights into WMMA B-fragment order (bf16).
//   QKVG blob: 64 ntiles x 8 ktiles, frag = 32 lanes x 16 bf16 (lane-major)
//   Wf   blob: 16 ntiles x 8 ktiles, same layout (128 KB total -> TDM to LDS)
// B-fragment mapping (K32 x N16): lane = n + 16*((k_local/8)&1), elem = k_local%8 + 8*(k_local>=16)
// ---------------------------------------------------------------------------
__global__ void k_pack(const float* __restrict__ Wq, const float* __restrict__ Wk,
                       const float* __restrict__ Wv, const float* __restrict__ Wg,
                       const float* __restrict__ Wf,
                       __bf16* __restrict__ packQKVG, __bf16* __restrict__ packF) {
    int tid = blockIdx.x * blockDim.x + threadIdx.x;
    if (tid < 64 * 8 * 512) {
        int e    = tid & 15;
        int lane = (tid >> 4) & 31;
        int kt   = (tid >> 9) & 7;
        int nt   = tid >> 12;                 // 0..63
        int klo  = (e & 7) + 8 * (lane >> 4) + 16 * (e >> 3);      // 0..31
        int n    = lane & 15;
        int oc   = nt * 16 + n;               // global out channel 0..1023
        int sel  = oc >> 8;                   // 0=Q 1=K 2=V 3=G
        int col  = oc & 255;
        const float* W = (sel == 0) ? Wq : (sel == 1) ? Wk : (sel == 2) ? Wv : Wg;
        packQKVG[tid] = f2bf(W[(kt * 32 + klo) * CC + col]);
    } else {
        int t = tid - 64 * 8 * 512;
        if (t < 16 * 8 * 512) {
            int e    = t & 15;
            int lane = (t >> 4) & 31;
            int kt   = (t >> 9) & 7;
            int nt   = t >> 12;               // 0..15
            int klo  = (e & 7) + 8 * (lane >> 4) + 16 * (e >> 3);
            int n    = lane & 15;
            packF[t] = f2bf(Wf[(kt * 32 + klo) * CC + nt * 16 + n]);
        }
    }
}

// ---------------------------------------------------------------------------
// Kernel A: fused LayerNorm + QKVG projection.
// One wave owns 16 tokens (all within one i-slice). Per output tile: one
// 16-load clause of B fragments, then 8 WMMAs with staggered loadcnt waits
// (no cross-tile double-buffer -> no scratch spills).
// Outputs: Q,K,G as [ih][n][d] bf16 ; V transposed as [ih][d][n] bf16.
// ---------------------------------------------------------------------------
__global__ void k_ln_proj(const float* __restrict__ z,
                          const float* __restrict__ lnw, const float* __restrict__ lnb,
                          const float* __restrict__ bg,
                          const __bf16* __restrict__ packW,
                          __bf16* __restrict__ Qb, __bf16* __restrict__ Kb,
                          __bf16* __restrict__ Vt, __bf16* __restrict__ Gb) {
    __shared__ __align__(32) __bf16 sZ[8][16][CC];   // 64 KB
    const int wave = threadIdx.x >> 5;
    const int lane = threadIdx.x & 31;
    const int wgid = blockIdx.x * 8 + wave;          // 0..4095
    const int tok0 = wgid * 16;
    const int c0   = lane * 8;

    v4f w0 = *(const v4f*)(lnw + c0);
    v4f w1 = *(const v4f*)(lnw + c0 + 4);
    v4f b0 = *(const v4f*)(lnb + c0);
    v4f b1 = *(const v4f*)(lnb + c0 + 4);

    for (int t = 0; t < 16; ++t) {
        const float* row = z + (size_t)(tok0 + t) * CC;
        v4f x0 = *(const v4f*)(row + c0);
        v4f x1 = *(const v4f*)(row + c0 + 4);
        float s = 0.f, ss = 0.f;
#pragma unroll
        for (int e = 0; e < 4; ++e) { s += x0[e]; ss += x0[e] * x0[e]; }
#pragma unroll
        for (int e = 0; e < 4; ++e) { s += x1[e]; ss += x1[e] * x1[e]; }
#pragma unroll
        for (int mk = 16; mk >= 1; mk >>= 1) {
            s  += __shfl_xor(s, mk, 32);
            ss += __shfl_xor(ss, mk, 32);
        }
        float mu  = s * (1.f / 256.f);
        float var = ss * (1.f / 256.f) - mu * mu;
        float inv = rsqrtf(var + 1e-5f);
#pragma unroll
        for (int e = 0; e < 4; ++e) {
            sZ[wave][t][c0 + e]     = f2bf((x0[e] - mu) * inv * w0[e] + b0[e]);
            sZ[wave][t][c0 + 4 + e] = f2bf((x1[e] - mu) * inv * w1[e] + b1[e]);
        }
    }
    __asm__ volatile("" ::: "memory");   // LDS in-order per wave

    const int m  = lane & 15;
    const int hi = lane >> 4;
    AFrag a[8];
#pragma unroll
    for (int kt = 0; kt < 8; ++kt) {
        a[kt].h[0] = *(const v8bf*)&sZ[wave][m][kt * 32 + hi * 8];
        a[kt].h[1] = *(const v8bf*)&sZ[wave][m][kt * 32 + 16 + hi * 8];
    }

    const int iS = tok0 >> 8;          // i slice (constant for the 16 tokens)
    const int n0 = (tok0 & 255) + 8 * hi;

    for (int nt = 0; nt < 64; ++nt) {
        // one clause of 16 global_load_b128, then WMMAs with staggered waits
        v16bf b[8];
#pragma unroll
        for (int kt = 0; kt < 8; ++kt)
            b[kt] = *(const v16bf*)(packW + (((nt * 8 + kt) * 32 + lane) << 4));
        v8f acc = {};
#pragma unroll
        for (int kt = 0; kt < 8; ++kt) acc = wmma_bf16(a[kt].v, b[kt], acc);

        const int sel = nt >> 4;                 // 0=Q 1=K 2=V 3=G
        const int h   = (nt >> 1) & 7;
        const int d   = (nt & 1) * 16 + m;
        const size_t ih = (size_t)(iS * HH + h);
        if (sel == 2) {
            __bf16* p = Vt + (ih * DD + d) * NN + n0;
#pragma unroll
            for (int j = 0; j < 8; ++j) p[j] = f2bf(acc[j]);
        } else {
            __bf16* base = (sel == 0) ? Qb : (sel == 1) ? Kb : Gb;
            float bias = (sel == 3) ? bg[(nt & 15) * 16 + m] : 0.f;
            __bf16* p = base + (ih * NN + n0) * DD + d;
#pragma unroll
            for (int j = 0; j < 8; ++j) p[(size_t)j * DD] = f2bf(acc[j] + bias);
        }
    }
}

// ---------------------------------------------------------------------------
// Kernel B: attention per (i,h). One wave = 16 query rows x all 256 keys.
// ---------------------------------------------------------------------------
__global__ void k_attn(const __bf16* __restrict__ Qb, const __bf16* __restrict__ Kb,
                       const __bf16* __restrict__ Vt, const __bf16* __restrict__ Gb,
                       const float* __restrict__ maskf, __bf16* __restrict__ X) {
    __shared__ __align__(32) __bf16 sW[8][16][NN];   // 64 KB softmax-weight staging
    const int wave = threadIdx.x >> 5;
    const int lane = threadIdx.x & 31;
    const int gw   = blockIdx.x * 8 + wave;          // 0..32767
    const int qt   = gw & 15;
    const int ih   = gw >> 4;                        // i*8 + h
    const int qbase = qt * 16;
    const int m  = lane & 15;
    const int hi = lane >> 4;

    const __bf16* Qp = Qb + (size_t)ih * NN * DD;
    const __bf16* Kp = Kb + (size_t)ih * NN * DD;
    const __bf16* Vp = Vt + (size_t)ih * DD * NN;
    const __bf16* Gp = Gb + (size_t)ih * NN * DD;
    const float*  mp = maskf + ((size_t)ih * NN + qbase) * NN;

    AFrag aq;
    aq.h[0] = *(const v8bf*)(Qp + (qbase + m) * DD + hi * 16);
    aq.h[1] = *(const v8bf*)(Qp + (qbase + m) * DD + hi * 16 + 8);

    const float scale = 0.1767766952966369f;         // 1/sqrt(32)
    v8f L[16];
#pragma unroll
    for (int kt = 0; kt < 16; ++kt) {
        if (kt < 15) __builtin_prefetch(mp + kt * 16 + 16, 0, 0);   // global_prefetch
        v16bf bk = *(const v16bf*)(Kp + (kt * 16 + m) * DD + hi * 16);
        v8f c = {};
        c = wmma_bf16(aq.v, bk, c);
#pragma unroll
        for (int j = 0; j < 8; ++j)
            c[j] = c[j] * scale + mp[(j + 8 * hi) * NN + kt * 16 + m];
        L[kt] = c;
    }

    // row softmax (row = j + 8*hi stays inside one 16-lane half -> xor 1/2/4/8)
#pragma unroll
    for (int j = 0; j < 8; ++j) {
        float mx = -3.0e38f;
#pragma unroll
        for (int kt = 0; kt < 16; ++kt) mx = fmaxf(mx, L[kt][j]);
#pragma unroll
        for (int s = 1; s < 16; s <<= 1) mx = fmaxf(mx, __shfl_xor(mx, s, 32));
        float sum = 0.f;
#pragma unroll
        for (int kt = 0; kt < 16; ++kt) {
            float e = __expf(L[kt][j] - mx);
            L[kt][j] = e;
            sum += e;
        }
#pragma unroll
        for (int s = 1; s < 16; s <<= 1) sum += __shfl_xor(sum, s, 32);
        float r = 1.f / sum;
#pragma unroll
        for (int kt = 0; kt < 16; ++kt)
            sW[wave][j + 8 * hi][kt * 16 + m] = f2bf(L[kt][j] * r);
    }
    __asm__ volatile("" ::: "memory");   // wave-private LDS region

    v8f acc[2] = {v8f{}, v8f{}};
#pragma unroll
    for (int kc = 0; kc < 8; ++kc) {
        AFrag aw;
        aw.h[0] = *(const v8bf*)&sW[wave][m][kc * 32 + hi * 16];
        aw.h[1] = *(const v8bf*)&sW[wave][m][kc * 32 + hi * 16 + 8];
        v16bf b0 = *(const v16bf*)(Vp + (0 * 16 + m) * NN + kc * 32 + hi * 16);
        v16bf b1 = *(const v16bf*)(Vp + (1 * 16 + m) * NN + kc * 32 + hi * 16);
        acc[0] = wmma_bf16(aw.v, b0, acc[0]);
        acc[1] = wmma_bf16(aw.v, b1, acc[1]);
    }

    const int i = ih >> 3;
    const int h = ih & 7;
#pragma unroll
    for (int nd = 0; nd < 2; ++nd) {
#pragma unroll
        for (int j = 0; j < 8; ++j) {
            int qrow = qbase + j + 8 * hi;
            float gl = bf2f(Gp[qrow * DD + nd * 16 + m]);
            float g  = 1.f / (1.f + __expf(-gl));
            float v  = g * acc[nd][j];
            X[(size_t)(i * NN + qrow) * CC + h * DD + nd * 16 + m] = f2bf(v);
        }
    }
}

// ---------------------------------------------------------------------------
// Kernel C: out = (X @ Wf + bf) * z_mask   (fp32 output)
// Wf fragments (exactly 128 KB) are staged into LDS with ONE Tensor-Data-Mover
// descriptor per block (wave 0 issues tensor_load_to_lds, waits TENSORcnt,
// then all waves consume B-fragments from LDS).
// ---------------------------------------------------------------------------
__global__ void k_out(const __bf16* __restrict__ X, const __bf16* __restrict__ packF,
                      const float* __restrict__ bias, const unsigned char* __restrict__ zmask,
                      float* __restrict__ out) {
    __shared__ __align__(32) __bf16 sF[16 * 8 * 512];   // 128 KB: all Wf fragments
    const int wave = threadIdx.x >> 5;
    const int lane = threadIdx.x & 31;

    if (threadIdx.x < 32) {
        // Tensor DMA descriptor (D#): 2D tensor, data_size=2B,
        // 512 elems/row x 128 rows = 128 KB, tile == tensor, global->LDS.
        unsigned long long ga = (unsigned long long)(const void*)packF;
        unsigned lds = (unsigned)(unsigned long long)(void*)&sF[0];
        u32x4 g0;
        g0[0] = 1u;                                    // count=1 (user, load)
        g0[1] = lds;                                   // lds_addr (bytes)
        g0[2] = (unsigned)(ga & 0xffffffffu);          // global_addr[31:0]
        g0[3] = (unsigned)((ga >> 32) & 0x01ffffffu)   // global_addr[56:32]
              | (2u << 30);                            // type = 2 ("image")
        i32x8 g1;
        g1[0] = (int)(1u << 16);                       // data_size = 1 -> 2 bytes
        g1[1] = (int)(512u << 16);                     // tensor_dim0 = 512
        g1[2] = (int)(128u << 16);                     // tensor_dim1 = 128
        g1[3] = (int)(512u << 16);                     // tile_dim0 = 512
        g1[4] = 128;                                   // tile_dim1 = 128
        g1[5] = 512;                                   // tensor_dim0_stride = 512
        g1[6] = 0;
        g1[7] = 0;
        i32x4 g2 = {0, 0, 0, 0};                       // 2D: groups 2/3 unused
        i32x4 g3 = {0, 0, 0, 0};
        i32x8 g4 = {0, 0, 0, 0, 0, 0, 0, 0};           // extra group (clang-23 6-arg form)
        __builtin_amdgcn_tensor_load_to_lds(g0, g1, g2, g3, g4, 0);
        __builtin_amdgcn_s_wait_tensorcnt(0);
    }
    __syncthreads();

    const int gw   = blockIdx.x * 8 + wave;          // 0..4095
    const int tok0 = gw * 16;
    const int m  = lane & 15;
    const int hi = lane >> 4;

    AFrag a[8];
#pragma unroll
    for (int kt = 0; kt < 8; ++kt) {
        a[kt].h[0] = *(const v8bf*)(X + (size_t)(tok0 + m) * CC + kt * 32 + hi * 8);
        a[kt].h[1] = *(const v8bf*)(X + (size_t)(tok0 + m) * CC + kt * 32 + 16 + hi * 8);
    }

    for (int nt = 0; nt < 16; ++nt) {
        v8f acc = {};
#pragma unroll
        for (int kt = 0; kt < 8; ++kt) {
            v16bf b = *(const v16bf*)(sF + (((nt * 8 + kt) * 32 + lane) << 4));
            acc = wmma_bf16(a[kt].v, b, acc);
        }
        float bs = bias[nt * 16 + m];
        float* po = out + (size_t)(tok0 + 8 * hi) * CC + nt * 16 + m;
        const unsigned char* pm = zmask + tok0 + 8 * hi;
#pragma unroll
        for (int j = 0; j < 8; ++j) {
            float mv = pm[j] ? 1.f : 0.f;
            po[(size_t)j * CC] = (acc[j] + bs) * mv;
        }
    }
}

// ---------------------------------------------------------------------------
// Launch
// ---------------------------------------------------------------------------
extern "C" void kernel_launch(void* const* d_in, const int* in_sizes, int n_in,
                              void* d_out, int out_size, void* d_ws, size_t ws_size,
                              hipStream_t stream) {
    (void)in_sizes; (void)n_in; (void)out_size; (void)ws_size;
    const float* z   = (const float*)d_in[0];
    const float* zmf = (const float*)d_in[1];
    const unsigned char* zm = (const unsigned char*)d_in[2];
    const float* lnw = (const float*)d_in[3];
    const float* lnb = (const float*)d_in[4];
    const float* Wq  = (const float*)d_in[5];
    const float* Wk  = (const float*)d_in[6];
    const float* Wv  = (const float*)d_in[7];
    const float* Wg  = (const float*)d_in[8];
    const float* bg  = (const float*)d_in[9];
    const float* Wf  = (const float*)d_in[10];
    const float* bf_ = (const float*)d_in[11];
    float* out = (float*)d_out;

    char* ws = (char*)d_ws;
    __bf16* packQKVG = (__bf16*)(ws);                       // 512 KB
    __bf16* packF    = (__bf16*)(ws + (512 << 10));         // 128 KB
    const size_t big = (size_t)2048 * NN * DD * 2;          // 33.5 MB each
    __bf16* Qb = (__bf16*)(ws + (640 << 10));
    __bf16* Kb = (__bf16*)((char*)Qb + big);
    __bf16* Vt = (__bf16*)((char*)Kb + big);
    __bf16* Gb = (__bf16*)((char*)Vt + big);
    __bf16* Xb = (__bf16*)((char*)Gb + big);

    k_pack   <<<1280, 256, 0, stream>>>(Wq, Wk, Wv, Wg, Wf, packQKVG, packF);
    k_ln_proj<<<512,  256, 0, stream>>>(z, lnw, lnb, bg, packQKVG, Qb, Kb, Vt, Gb);
    k_attn   <<<4096, 256, 0, stream>>>(Qb, Kb, Vt, Gb, zmf, Xb);
    k_out    <<<512,  256, 0, stream>>>(Xb, packF, bf_, zm, out);
}
